// GateNetwork_Gobal_68659347194405
// MI455X (gfx1250) — compile-verified, matrix operand order checked by reading, not run on hardware
//
#include <hip/hip_runtime.h>
#include <cstdint>
#include <cstddef>

// Problem constants (match reference)
#define NB    16384   // batch rows
#define ND    8192    // feature dim (K)
#define NE    128     // experts (N)
#define TOPK  8

// Tiling
#define KC        512            // K-chunk staged in LDS per iteration
#define NSTEP     (KC / 32)      // WMMA K-steps per chunk
#define ROWS      32             // rows per workgroup (2 row-tiles of 16)
#define A_STRIDE  260            // dwords per LDS row: 256 bf16-pairs + 4 pad (bank-conflict free)
#define S_STRIDE  132            // padded score row stride (floats)

typedef __attribute__((ext_vector_type(16))) __bf16 v16bf;
typedef __attribute__((ext_vector_type(8)))  float  v8f;
typedef __attribute__((ext_vector_type(4)))  float  f32x4;   // native vec for NT loads

union BFrag {
    uint32_t u[8];
    v16bf    v;
};

// Round-to-nearest-even f32 -> bf16, packed pair (lo = a, hi = b)
__device__ __forceinline__ uint32_t pack_bf16(float a, float b) {
    uint32_t ua = __builtin_bit_cast(uint32_t, a);
    uint32_t ub = __builtin_bit_cast(uint32_t, b);
    ua = (ua + 0x7FFFu + ((ua >> 16) & 1u)) >> 16;
    ub = (ub + 0x7FFFu + ((ub >> 16) & 1u)) & 0xFFFF0000u;
    return ua | ub;
}

// Load an 8-dword fragment as two b128s (works for LDS and global pointers).
__device__ __forceinline__ void load_frag(BFrag& f, const uint32_t* base) {
    *reinterpret_cast<uint4*>(&f.u[0]) = *reinterpret_cast<const uint4*>(base);
    *reinterpret_cast<uint4*>(&f.u[4]) = *reinterpret_cast<const uint4*>(base + 8);
}

// ---------------------------------------------------------------------------
// Kernel 1: pack W [NE][ND] f32 (row-major) -> bf16 pairs [NE][ND/2] in d_ws.
// 2 MB result lives in L2 for the whole GEMM. Re-run every launch (stateless).
// ---------------------------------------------------------------------------
__global__ __launch_bounds__(256)
void w_pack_bf16(const float* __restrict__ W, uint32_t* __restrict__ Wp) {
    int i = blockIdx.x * 256 + threadIdx.x;            // pair index
    float2 f = reinterpret_cast<const float2*>(W)[i];
    Wp[i] = pack_bf16(f.x, f.y);
}

// ---------------------------------------------------------------------------
// Kernel 2: fused gating GEMM (bf16 WMMA) + softmax + top-8 + re-softmax.
// Workgroup: 256 threads = 8 waves. Wave w owns experts [16w, 16w+16).
// Each wave computes two 16x16 tiles (rows 0-15 and 16-31 of the block).
// Inner loop is software-pipelined: fragment loads for step s+1 are issued
// before the WMMAs of step s (register double buffers).
// ---------------------------------------------------------------------------
__global__ __launch_bounds__(256)
void gate_topk_kernel(const float*    __restrict__ x,
                      const uint32_t* __restrict__ Wp,   // bf16-packed W
                      const float*    __restrict__ bias,
                      float*          __restrict__ out_p,
                      int*            __restrict__ out_i)
{
    __shared__ uint32_t aTile[ROWS * A_STRIDE];          // x chunk as bf16 pairs
    __shared__ float    sc[ROWS * S_STRIDE];             // scores [row][expert]

    const int tid  = threadIdx.x;
    const int wave = tid >> 5;                           // 0..7
    const int lane = tid & 31;
    const int g    = lane >> 4;                          // K-half select
    const int ln   = lane & 15;                          // row / expert-in-tile
    const int rowBase = blockIdx.x * ROWS;
    const int expert  = wave * 16 + ln;

    const float     bval = bias[expert];
    const uint32_t* wrow = Wp + (size_t)expert * (ND / 2);

    v8f acc0 = {};   // rows 0..15 of block
    v8f acc1 = {};   // rows 16..31 of block

    for (int k0 = 0; k0 < ND; k0 += KC) {
        // ---- Stage x[rowBase:+ROWS, k0:+KC] -> LDS as packed bf16 pairs ----
        // ROWS*KC/4 = 4096 float4 slots, 16 per thread, coalesced per row.
        // x is streamed exactly once -> non-temporal loads (don't pollute L2).
        #pragma unroll
        for (int it = 0; it < (ROWS * KC / 4) / 256; ++it) {
            int f  = it * 256 + tid;
            int r  = f >> 7;                              // KC/4 = 128 float4 per row
            int c4 = f & 127;
            const f32x4* src = reinterpret_cast<const f32x4*>(
                x + (size_t)(rowBase + r) * ND + k0 + 4 * c4);
            f32x4 v4 = __builtin_nontemporal_load(src);
            uint2 p;
            p.x = pack_bf16(v4[0], v4[1]);
            p.y = pack_bf16(v4[2], v4[3]);
            *reinterpret_cast<uint2*>(&aTile[r * A_STRIDE + 2 * c4]) = p;
        }
        __syncthreads();

        // ---- Prefetch next x chunk into L2 while we compute this one ----
        // 512 cachelines (32 rows x 16 x 128B) per chunk, 2 per thread.
        if (k0 + KC < ND) {
            #pragma unroll
            for (int it = 0; it < 2; ++it) {
                int l   = it * 256 + tid;
                int r   = l >> 4;                         // 16 lines per row
                int off = (l & 15) << 5;                  // *32 floats = 128B
                __builtin_prefetch(
                    x + (size_t)(rowBase + r) * ND + (k0 + KC) + off, 0, 0);
            }
        }

        // ---- Software-pipelined WMMA over NSTEP K-steps ----
        BFrag bf[2], a0[2], a1[2];
        {   // preload step 0
            const uint32_t* wb = wrow + (k0 >> 1) + 4 * g;
            load_frag(bf[0], wb);
            const int pb = 4 * g;
            load_frag(a0[0], &aTile[ln * A_STRIDE + pb]);
            load_frag(a1[0], &aTile[(ln + 16) * A_STRIDE + pb]);
        }
        #pragma unroll
        for (int s = 0; s < NSTEP; ++s) {
            const int cur = s & 1;
            const int nxt = cur ^ 1;
            if (s + 1 < NSTEP) {
                // Issue step s+1 loads BEFORE the WMMAs of step s.
                const uint32_t* wb = wrow + ((k0 + 32 * (s + 1)) >> 1) + 4 * g;
                load_frag(bf[nxt], wb);
                const int pb = 16 * (s + 1) + 4 * g;
                load_frag(a0[nxt], &aTile[ln * A_STRIDE + pb]);
                load_frag(a1[nxt], &aTile[(ln + 16) * A_STRIDE + pb]);
            }
            acc0 = __builtin_amdgcn_wmma_f32_16x16x32_bf16(
                false, a0[cur].v, false, bf[cur].v, (short)0, acc0, false, false);
            acc1 = __builtin_amdgcn_wmma_f32_16x16x32_bf16(
                false, a1[cur].v, false, bf[cur].v, (short)0, acc1, false, false);
        }
        __syncthreads();
    }

    // ---- Scatter scores (+bias) to LDS: C VGPR r -> row r + 8g ----
    #pragma unroll
    for (int r = 0; r < 8; ++r) {
        sc[(r + 8 * g) * S_STRIDE + expert]      = acc0[r] + bval;
        sc[(16 + r + 8 * g) * S_STRIDE + expert] = acc1[r] + bval;
    }
    __syncthreads();

    // ---- Per-row softmax -> top-8 -> softmax over top-8 probs ----
    if (tid < ROWS) {
        const float* s = &sc[tid * S_STRIDE];

        float mx = -INFINITY;
        for (int e = 0; e < NE; ++e) mx = fmaxf(mx, s[e]);
        float sum = 0.f;
        for (int e = 0; e < NE; ++e) sum += __expf(s[e] - mx);
        float inv = 1.0f / sum;

        // Register-resident descending top-8 (branchless bubble insert;
        // strict '>' keeps lower index first on ties, like lax.top_k).
        float tv[TOPK]; int ti[TOPK];
        #pragma unroll
        for (int k = 0; k < TOPK; ++k) { tv[k] = -INFINITY; ti[k] = -1; }
        for (int e = 0; e < NE; ++e) {
            float v = s[e];
            int   idx = e;
            #pragma unroll
            for (int k = 0; k < TOPK; ++k) {
                bool  gt = v > tv[k];
                float nv = gt ? v   : tv[k];
                int   ni = gt ? idx : ti[k];
                v   = gt ? tv[k] : v;
                idx = gt ? ti[k] : idx;
                tv[k] = nv; ti[k] = ni;
            }
        }

        // top-k probabilities from the first softmax
        float p[TOPK];
        #pragma unroll
        for (int k = 0; k < TOPK; ++k) p[k] = __expf(tv[k] - mx) * inv;

        // reference applies a SECOND softmax over those probabilities
        float m2 = p[0], s2 = 0.f;               // p[] sorted descending
        #pragma unroll
        for (int k = 0; k < TOPK; ++k) { p[k] = __expf(p[k] - m2); s2 += p[k]; }
        float i2 = 1.0f / s2;

        const int row = rowBase + tid;
        #pragma unroll
        for (int k = 0; k < TOPK; ++k) {
            out_p[row * TOPK + k] = p[k] * i2;
            out_i[row * TOPK + k] = ti[k];
        }
    }
}

// ---------------------------------------------------------------------------
extern "C" void kernel_launch(void* const* d_in, const int* in_sizes, int n_in,
                              void* d_out, int out_size, void* d_ws, size_t ws_size,
                              hipStream_t stream) {
    const float* x = (const float*)d_in[0];   // [NB, ND] f32
    const float* W = (const float*)d_in[1];   // [NE, ND] f32
    const float* b = (const float*)d_in[2];   // [NE] f32

    uint32_t* Wp   = (uint32_t*)d_ws;                       // 2 MB bf16-packed W
    float*    outp = (float*)d_out;                         // [NB, 8] probs
    int*      outi = (int*)d_out + (size_t)NB * TOPK;       // [NB, 8] indices

    w_pack_bf16<<<(NE * ND / 2) / 256, 256, 0, stream>>>(W, Wp);
    gate_topk_kernel<<<NB / ROWS, 256, 0, stream>>>(x, Wp, b, outp, outi);
}
